// SupConLoss_73418170958288
// MI455X (gfx1250) — matrix operand order repaired and checked by reading, not compiled, not run on hardware
//
#include <hip/hip_runtime.h>
#include <hip/hip_bf16.h>

typedef __attribute__((ext_vector_type(16))) __bf16 v16bf;
typedef __attribute__((ext_vector_type(8)))  float  v8f;

constexpr int D     = 1024;   // feature dim
constexpr int HALF  = 2048;   // B (batch)
constexpr int NROWS = 4096;   // 2B rows of Z
constexpr float INV_T = 1.0f / 0.07f;

union Frag { v16bf v; uint4 q[2]; };                 // 32 bytes: 16 bf16 per lane
union Bf16Bits { __hip_bfloat16 h; unsigned short u; };

// ---------------------------------------------------------------------------
// 1) L2-normalize each of the 4096 vectors, emit bf16 Z[4096][1024].
//    rows [0,2048) = features[:,0,:], rows [2048,4096) = features[:,1,:]
// ---------------------------------------------------------------------------
__global__ __launch_bounds__(256)
void normalize_kernel(const float* __restrict__ feat, unsigned short* __restrict__ Z) {
    __shared__ float red[256];
    const int row = blockIdx.x;
    const int b = row & (HALF - 1);
    const int s = row >> 11;
    const float* v = feat + ((size_t)b * 2 + s) * D;

    float x[4];
    float ss = 0.f;
#pragma unroll
    for (int j = 0; j < 4; ++j) {
        x[j] = v[threadIdx.x + j * 256];
        ss += x[j] * x[j];
    }
    red[threadIdx.x] = ss;
    __syncthreads();
    for (int off = 128; off > 0; off >>= 1) {
        if (threadIdx.x < off) red[threadIdx.x] += red[threadIdx.x + off];
        __syncthreads();
    }
    const float inv = 1.0f / fmaxf(sqrtf(red[0]), 1e-12f);
#pragma unroll
    for (int j = 0; j < 4; ++j) {
        Bf16Bits cv;
        cv.h = __float2bfloat16(x[j] * inv);
        Z[(size_t)row * D + threadIdx.x + j * 256] = cv.u;
    }
}

// ---------------------------------------------------------------------------
// 2) zero denom/pos accumulators (must be re-zeroed every call)
// ---------------------------------------------------------------------------
__global__ void zero_kernel(float* __restrict__ p, int n) {
    int i = blockIdx.x * blockDim.x + threadIdx.x;
    if (i < n) p[i] = 0.f;
}

// ---------------------------------------------------------------------------
// 3) Fused symmetric GEMM (Z Z^T) + exp-mask-rowsum epilogue.
//    128x128 tile per block, upper-triangular blocks only (bx >= by).
//    8 waves arranged 4(M) x 2(N); each wave: 2x4 wmma_f32_16x16x32_bf16 accs.
// ---------------------------------------------------------------------------
__global__ __launch_bounds__(256)
void simexp_kernel(const unsigned short* __restrict__ Z,
                   float* __restrict__ denom,
                   float* __restrict__ pos) {
    const int bx = blockIdx.x, by = blockIdx.y;
    if (bx < by) return;                       // symmetric: upper triangle only
    const bool diag = (bx == by);

    const int blockM = by * 128;
    const int blockN = bx * 128;
    const int wave  = threadIdx.x >> 5;
    const int waveM = wave >> 1;               // 0..3  (32 rows each)
    const int waveN = wave & 1;                // 0..1  (64 cols each)
    const int lane  = threadIdx.x & 31;
    const int halfl = lane >> 4;               // lane group 0 / 1
    const int l15   = lane & 15;

    const int rowA0 = blockM + waveM * 32 + l15;   // + m*16
    const int colB0 = blockN + waveN * 64 + l15;   // + n*16

    v8f acc[2][4] = {};

    for (int k0 = 0; k0 < D; k0 += 32) {
        Frag A[2], Bm[4];
        // A (16x32 bf16): lane l<16 holds K {k0..k0+7, k0+16..k0+23} of row M=l15
        //                 lane l>=16 holds K {+8..+15, +24..+31}   -> 2 x b128
#pragma unroll
        for (int m = 0; m < 2; ++m) {
            const unsigned short* p = Z + (size_t)(rowA0 + m * 16) * D + k0 + halfl * 8;
            A[m].q[0] = *(const uint4*)(p);
            A[m].q[1] = *(const uint4*)(p + 16);
        }
        // B (32x16 bf16): lane holds column N=l15, K {k0+halfl*8..+7, k0+16+halfl*8..+7}
        //                 (same K order as A) -> 2 x b128
#pragma unroll
        for (int n = 0; n < 4; ++n) {
            const unsigned short* p = Z + (size_t)(colB0 + n * 16) * D + k0 + halfl * 8;
            Bm[n].q[0] = *(const uint4*)(p);
            Bm[n].q[1] = *(const uint4*)(p + 16);
        }
#pragma unroll
        for (int m = 0; m < 2; ++m)
#pragma unroll
            for (int n = 0; n < 4; ++n)
                acc[m][n] = __builtin_amdgcn_wmma_f32_16x16x32_bf16(
                    false, A[m].v, false, Bm[n].v,
                    (short)0, acc[m][n], false, false);
    }

    // ---- epilogue: exp(sim/T), self-mask, row sums (+ col sums off-diagonal)
    // C layout: VGPR e -> row m*16 + e + 8*halfl ; lane l15 -> col n*16 + l15
#pragma unroll
    for (int m = 0; m < 2; ++m) {
        const int rowBase = blockM + waveM * 32 + m * 16 + halfl * 8;
        float rs[8];
        float cs[4];
#pragma unroll
        for (int e = 0; e < 8; ++e) rs[e] = 0.f;
#pragma unroll
        for (int n = 0; n < 4; ++n) cs[n] = 0.f;

#pragma unroll
        for (int n = 0; n < 4; ++n) {
            const int col = blockN + waveN * 64 + n * 16 + l15;
#pragma unroll
            for (int e = 0; e < 8; ++e) {
                const int row = rowBase + e;
                const float s = acc[m][n][e];
                // positive pair: col == row +/- B (only in off-diag blocks)
                if (col == ((row < HALF) ? row + HALF : row - HALF)) {
                    pos[row] = s;
                    pos[col] = s;
                }
                const float ev = (row == col) ? 0.f : __expf(s * INV_T);
                rs[e] += ev;
                cs[n] += ev;
            }
        }
        // row sums: butterfly across the 16 lanes of each half-group
#pragma unroll
        for (int off = 1; off < 16; off <<= 1)
#pragma unroll
            for (int e = 0; e < 8; ++e)
                rs[e] += __shfl_xor(rs[e], off, 32);
        if (l15 == 0) {
#pragma unroll
            for (int e = 0; e < 8; ++e)
                atomicAdd(&denom[rowBase + e], rs[e]);
        }
        // col sums (symmetry scatter) for strictly off-diagonal blocks
        if (!diag) {
#pragma unroll
            for (int n = 0; n < 4; ++n) {
                float c = cs[n] + __shfl_xor(cs[n], 16, 32);
                if (halfl == 0)
                    atomicAdd(&denom[blockN + waveN * 64 + n * 16 + l15], c);
            }
        }
    }
}

// ---------------------------------------------------------------------------
// 4) loss = mean_i( log(denom_i) - pos_i / T )
// ---------------------------------------------------------------------------
__global__ __launch_bounds__(256)
void finalize_kernel(const float* __restrict__ denom,
                     const float* __restrict__ pos,
                     float* __restrict__ out) {
    __shared__ float red[256];
    float s = 0.f;
    for (int i = threadIdx.x; i < NROWS; i += 256)
        s += (logf(denom[i]) - pos[i] * INV_T);
    red[threadIdx.x] = s;
    __syncthreads();
    for (int off = 128; off > 0; off >>= 1) {
        if (threadIdx.x < off) red[threadIdx.x] += red[threadIdx.x + off];
        __syncthreads();
    }
    if (threadIdx.x == 0) out[0] = red[0] / (float)NROWS;
}

// ---------------------------------------------------------------------------
extern "C" void kernel_launch(void* const* d_in, const int* in_sizes, int n_in,
                              void* d_out, int out_size, void* d_ws, size_t ws_size,
                              hipStream_t stream) {
    const float* feat = (const float*)d_in[0];
    float* out = (float*)d_out;

    unsigned short* Z = (unsigned short*)d_ws;                       // 8 MB bf16
    float* denom = (float*)((char*)d_ws + (size_t)NROWS * D * sizeof(unsigned short));
    float* pos   = denom + NROWS;

    normalize_kernel<<<NROWS, 256, 0, stream>>>(feat, Z);
    zero_kernel<<<(2 * NROWS + 255) / 256, 256, 0, stream>>>(denom, 2 * NROWS);
    dim3 grid(32, 32);
    simexp_kernel<<<grid, 256, 0, stream>>>(Z, denom, pos);
    finalize_kernel<<<1, 256, 0, stream>>>(denom, pos, out);
}
